// CIF_38594576122088
// MI455X (gfx1250) — matrix operand order repaired and checked by reading, not compiled, not run on hardware
//
#include <hip/hip_runtime.h>
#include <hip/hip_bf16.h>
#include <math.h>

// ---------------------------------------------------------------------------
// CIF (continuous integrate-and-fire) for MI455X / gfx1250, wave32 + WMMA.
//
// alpha[b,t] = sigmoid( sum_{k,d} E[b,t+k-5,d]*weff[k,d] + beff ) with
// weff[k,d] = sum_c conv_w[k,d,c]*proj_w[c]  (rank-1 projection folded into
// the conv => 189 GFLOP -> ~1 GFLOP; bound by one 64MB read of eouts).
// Computed with v_wmma_f32_16x16x32_bf16, conv taps as the WMMA N dimension.
// ---------------------------------------------------------------------------

typedef __attribute__((ext_vector_type(16))) __bf16 v16bf;
typedef __attribute__((ext_vector_type(8)))  __bf16 v8bf;
typedef __attribute__((ext_vector_type(8)))  float  v8f;
typedef __attribute__((ext_vector_type(4)))  float  v4f;

#define BB   16
#define TT   2048
#define DD   512
#define CC   512
#define KW   11     // conv taps
#define KH   5      // half width

// ---------------------------------------------------------------------------
// Zero-fill helper (d_out regions and the zero-row scratch).
__global__ void k_zero(float* __restrict__ p, int n) {
  for (int i = blockIdx.x * blockDim.x + threadIdx.x; i < n;
       i += gridDim.x * blockDim.x)
    p[i] = 0.0f;
}

// ---------------------------------------------------------------------------
// Fold projection into conv weights (stored bf16 for direct WMMA B-fragments):
//   weff[k][d] = sum_c conv_w[k][d][c] * proj_w[c]   (rows k>=11 are zero)
//   beff       = sum_c conv_b[c] * proj_w[c] + proj_b
__global__ void k_weff(const float* __restrict__ conv_w,
                       const float* __restrict__ conv_b,
                       const float* __restrict__ proj_w,
                       const float* __restrict__ proj_b,
                       __bf16* __restrict__ weff,
                       float* __restrict__ beff) {
  int tid = blockIdx.x * blockDim.x + threadIdx.x;    // 0 .. 16*512-1
  if (tid >= 16 * DD) return;
  int k = tid >> 9, d = tid & (DD - 1);
  float s = 0.0f;
  if (k < KW) {
    const float* row = conv_w + ((size_t)(k * DD + d) << 9);  // [k][d][:] contiguous
    for (int c = 0; c < CC; ++c) s += row[c] * proj_w[c];
  }
  weff[tid] = (__bf16)s;
  if (tid == 0) {
    float bs = proj_b[0];
    for (int c = 0; c < CC; ++c) bs += conv_b[c] * proj_w[c];
    *beff = bs;
  }
}

// ---------------------------------------------------------------------------
// WMMA alpha kernel.  One wave per (b, t0) tile of 16 output times.
//  B [32x16]: column j = weff[j][kchunk] (bf16, j>=11 zero).
//  A tile0 rows t0-5 .. t0+10, A tile1 rows t0+11 .. t0+26; out-of-range rows
//  read a zero-filled scratch row (pointer select, no per-element cndmask).
//  C[i][j] of a tile with base r0 contributes to alpha_pre[r0 + i + 5 - j].
__global__ void __launch_bounds__(256)
k_alpha_wmma(const float* __restrict__ eouts,
             const __bf16* __restrict__ weff,
             const float* __restrict__ beff,
             const float* __restrict__ zrow,   // 512 zero floats
             float* __restrict__ alpha) {
  __shared__ float lds[8][2][16][16];

  const int lane = threadIdx.x & 31;
  const int wv   = threadIdx.x >> 5;
  const int tile = blockIdx.x * 8 + wv;          // 2048 tiles total
  const int b    = tile >> 7;                    // 128 tiles per batch
  const int t0   = (tile & 127) << 4;
  const int half = lane >> 4;                    // 0: lanes 0-15, 1: lanes 16-31
  const int ln   = lane & 15;
  const int aOff = half ? 8 : 0;                 // A K-offset per half-wave

  const float* ebase = eouts + ((size_t)b * TT) * DD;

  const int r0 = t0 - KH + ln;                   // tile0 row for this lane
  const int r1 = t0 + KW + ln;                   // tile1 row
  // Loop-invariant pointer select: invalid halo rows read the zero row.
  const float* row0 = (r0 >= 0) ? (ebase + (size_t)r0 * DD) : zrow;
  const float* row1 = (r1 < TT) ? (ebase + (size_t)r1 * DD) : zrow;
  const __bf16* wrow = weff + ln * DD + (half ? 8 : 0);

  v8f c0 = {0.f, 0.f, 0.f, 0.f, 0.f, 0.f, 0.f, 0.f};
  v8f c1 = {0.f, 0.f, 0.f, 0.f, 0.f, 0.f, 0.f, 0.f};

  for (int kc = 0; kc < DD / 32; ++kc) {
    const int kb = kc * 32;

    // --- B fragment: 2 x 16B bf16 loads, no conversion
    v8bf blo = *(const v8bf*)(wrow + kb);
    v8bf bhi = *(const v8bf*)(wrow + kb + 16);
    v16bf bm;
#pragma unroll
    for (int e = 0; e < 8; ++e) { bm[e] = blo[e]; bm[e + 8] = bhi[e]; }

    // --- A fragments: unconditional b128 loads + f32->bf16 converts
    const float* ap0 = row0 + kb + aOff;
    const float* ap1 = row1 + kb + aOff;
    v4f x00 = *(const v4f*)(ap0);      v4f x01 = *(const v4f*)(ap0 + 4);
    v4f y00 = *(const v4f*)(ap0 + 16); v4f y01 = *(const v4f*)(ap0 + 20);
    v4f x10 = *(const v4f*)(ap1);      v4f x11 = *(const v4f*)(ap1 + 4);
    v4f y10 = *(const v4f*)(ap1 + 16); v4f y11 = *(const v4f*)(ap1 + 20);
    v16bf a0, a1;
#pragma unroll
    for (int e = 0; e < 4; ++e) {
      a0[e]      = (__bf16)x00[e];  a0[e + 4]  = (__bf16)x01[e];
      a0[e + 8]  = (__bf16)y00[e];  a0[e + 12] = (__bf16)y01[e];
      a1[e]      = (__bf16)x10[e];  a1[e + 4]  = (__bf16)x11[e];
      a1[e + 8]  = (__bf16)y10[e];  a1[e + 12] = (__bf16)y11[e];
    }

    c0 = __builtin_amdgcn_wmma_f32_16x16x32_bf16(false, a0, false, bm,
                                                 (short)0, c0, false, false);
    c1 = __builtin_amdgcn_wmma_f32_16x16x32_bf16(false, a1, false, bm,
                                                 (short)0, c1, false, false);
  }

  // C layout: vgpr v, lane l -> M = v + 8*(l>>4), N = l&15
#pragma unroll
  for (int v = 0; v < 8; ++v) {
    lds[wv][0][v + 8 * half][ln] = c0[v];
    lds[wv][1][v + 8 * half][ln] = c1[v];
  }
  __syncthreads();

  if (lane < 16) {
    float s = *beff;
#pragma unroll
    for (int j = 0; j < KW; ++j) {
      int i = lane + j;                     // t = t0+lane needs C[i][j]
      s += (i < 16) ? lds[wv][0][i][j] : lds[wv][1][i - 16][j];
    }
    alpha[b * TT + t0 + lane] = 1.0f / (1.0f + __expf(-s));
  }
}

// ---------------------------------------------------------------------------
// Per-batch alpha sum (for normalization).
__global__ void k_asum(const float* __restrict__ alpha, float* __restrict__ asum) {
  __shared__ float red[256];
  int b = blockIdx.x;
  float s = 0.f;
  for (int t = threadIdx.x; t < TT; t += 256) s += alpha[b * TT + t];
  red[threadIdx.x] = s;
  __syncthreads();
  for (int st = 128; st > 0; st >>= 1) {
    if (threadIdx.x < st) red[threadIdx.x] += red[threadIdx.x + st];
    __syncthreads();
  }
  if (threadIdx.x == 0) asum[b] = red[0];
}

// ---------------------------------------------------------------------------
// Serial integrate-and-fire scan.  One wave; lane b = batch b.
// Emits aws[b][n][t] directly (w_old at n_tok, ak2 at n_tok+1 on fire),
// plus fire times for the segmented fired-row reduction.
__global__ void k_scan(const float* __restrict__ alpha,
                       const float* __restrict__ asum,
                       const int* __restrict__ elens,
                       const int* __restrict__ ylens,
                       float* __restrict__ aws,      // [B][(L+1)][T], pre-zeroed
                       int* __restrict__ ftime,      // [B][128]
                       int* __restrict__ nfired,     // [B]
                       int L) {
  int b = threadIdx.x;
  if (b >= BB) return;
  const int   elen  = elens[b];
  const int   ylen  = ylens[b];
  const float scale = (float)ylen / asum[b];
  const float* ap   = alpha + b * TT;
  float* awsRow     = aws + (size_t)b * (L + 1) * TT;

  float accum = 0.0f;
  int   ntok  = 0;
  for (int t = 0; t < TT; ++t) {
    float a = ap[t] * scale;
    accum += a;
    bool active = (t < elen) && (ntok < ylen);
    bool fire   = active && (accum >= 0.9f);
    float ak1 = 1.0f - accum;
    float ak2 = a - ak1;
    float w_old = fire ? ak1 : (active ? a : 0.0f);
    awsRow[ntok * TT + t] = w_old;
    if (fire) {
      awsRow[(ntok + 1) * TT + t] = ak2;
      ftime[(b << 7) + ntok] = t;
      ++ntok;
      accum = ak2;
    }
  }
  nfired[b] = ntok;
}

// ---------------------------------------------------------------------------
// fired[b][n][:] = sum_t aws[b][n][t] * eouts[b][t][:] over the token's
// segment [ftime[n-1], ftime[n]] (aws row n holds exactly the weights the
// reference's running "state" accumulates).  Rows that never fired stay zero.
__global__ void __launch_bounds__(256)
k_fired(const float* __restrict__ eouts,
        const float* __restrict__ aws,
        const int* __restrict__ ftime,
        const int* __restrict__ nfired,
        float* __restrict__ fired,
        int L) {
  int blk = blockIdx.x;
  int b = blk / L, n = blk % L;
  if (n >= nfired[b]) return;

  int tEnd   = ftime[(b << 7) + n];
  int tStart = (n > 0) ? ftime[(b << 7) + n - 1] : 0;  // prev fire t holds ak2 in row n

  const float* wrow  = aws + ((size_t)b * (L + 1) + n) * TT;
  const float* ebase = eouts + ((size_t)b * TT) * DD;
  const int d = threadIdx.x;

  float acc0 = 0.0f, acc1 = 0.0f;
  for (int t = tStart; t <= tEnd; ++t) {
    float w = wrow[t];
    const float* e = ebase + (size_t)t * DD;
    __builtin_prefetch(e + DD, 0, 1);
    acc0 += w * e[d];
    acc1 += w * e[d + 256];
  }
  float* out = fired + ((size_t)b * L + n) * DD;
  out[d] = acc0;
  out[d + 256] = acc1;
}

// ---------------------------------------------------------------------------
extern "C" void kernel_launch(void* const* d_in, const int* in_sizes, int n_in,
                              void* d_out, int out_size, void* d_ws, size_t ws_size,
                              hipStream_t stream) {
  (void)in_sizes; (void)n_in; (void)ws_size;
  const float* eouts  = (const float*)d_in[0];
  const float* conv_w = (const float*)d_in[1];
  const float* conv_b = (const float*)d_in[2];
  const float* proj_w = (const float*)d_in[3];
  const float* proj_b = (const float*)d_in[4];
  const int*   elens  = (const int*)d_in[5];
  const int*   ylens  = (const int*)d_in[6];

  // out_size = B*L*D + B*T + B*(L+1)*T = 40960*L + 65536  ->  recover L
  int L = (out_size - (BB * TT) * 2) / (BB * DD + BB * TT);
  if (L < 1) L = 1;

  char* ws = (char*)d_ws;
  __bf16* weff  = (__bf16*)(ws);           // 16*512 bf16 = 16 KB
  float*  beff  = (float*) (ws + 16384);
  float*  asum  = (float*) (ws + 16448);   // 16 f32
  int*    ftime = (int*)   (ws + 16512);   // 16*128 int
  int*    nfired= (int*)   (ws + 24704);   // 16 int
  float*  zrow  = (float*) (ws + 24768);   // 512 zero f32 (16B aligned)

  float* out_fired = (float*)d_out;                     // [B][L][D]
  float* out_alpha = out_fired + (size_t)BB * L * DD;   // [B][T]
  float* out_aws   = out_alpha + (size_t)BB * TT;       // [B][L+1][T]

  hipLaunchKernelGGL(k_zero, dim3(1024), dim3(256), 0, stream,
                     (float*)d_out, out_size);
  hipLaunchKernelGGL(k_zero, dim3(2), dim3(256), 0, stream, zrow, DD);
  hipLaunchKernelGGL(k_weff, dim3(32), dim3(256), 0, stream,
                     conv_w, conv_b, proj_w, proj_b, weff, beff);
  hipLaunchKernelGGL(k_alpha_wmma, dim3(256), dim3(256), 0, stream,
                     eouts, weff, beff, zrow, out_alpha);
  hipLaunchKernelGGL(k_asum, dim3(16), dim3(256), 0, stream, out_alpha, asum);
  hipLaunchKernelGGL(k_scan, dim3(1), dim3(32), 0, stream,
                     out_alpha, asum, elens, ylens, out_aws, ftime, nfired, L);
  hipLaunchKernelGGL(k_fired, dim3(BB * L), dim3(256), 0, stream,
                     eouts, out_aws, ftime, nfired, out_fired, L);
}
